// FractalRecurrentALU_20383914786848
// MI455X (gfx1250) — matrix operand order repaired and checked by reading, not compiled, not run on hardware
//
#include <hip/hip_runtime.h>
#include <hip/hip_bf16.h>

typedef __attribute__((ext_vector_type(16))) __bf16 v16bf;
typedef __attribute__((ext_vector_type(8)))  __bf16 v8bf;
typedef __attribute__((ext_vector_type(8)))  float  v8f;

#define Bsz   4096
#define Ssz   64
#define Dsz   512
#define DEPTH 4
#define ROWS  32          // batch rows per workgroup
#define THREADS 256       // 8 waves
#define LDX   (Dsz + 4)   // fp32 state stride (pad)
#define LDY   (Dsz + 8)   // bf16 buffer stride (pad, keeps 16B align)

static __device__ __forceinline__ __bf16 f2bf(float f) {
    union { float f; unsigned u; } a; a.f = f;
    unsigned r = a.u + 0x7FFFu + ((a.u >> 16) & 1u);   // round-to-nearest-even
    union { unsigned short s; __bf16 b; } o; o.s = (unsigned short)(r >> 16);
    return o.b;
}

// ---------------- weight prep: fp32 -> bf16, transposed to [N][K] ----------------
__global__ void prep_vwt(const float* __restrict__ qkv_w, __bf16* __restrict__ vwt) {
    int i = blockIdx.x * 256 + threadIdx.x;           // 4*512*512
    int k = i & 511, n = (i >> 9) & 511, l = i >> 18;
    vwt[i] = f2bf(qkv_w[(size_t)l * Dsz * 3 * Dsz + (size_t)k * 3 * Dsz + 2 * Dsz + n]);
}
__global__ void prep_owt(const float* __restrict__ out_w, __bf16* __restrict__ owt) {
    int i = blockIdx.x * 256 + threadIdx.x;           // 4*512*512
    int k = i & 511, n = (i >> 9) & 511, l = i >> 18;
    owt[i] = f2bf(out_w[(size_t)l * Dsz * Dsz + (size_t)k * Dsz + n]);
}
__global__ void prep_ff1t(const float* __restrict__ ff1_w, __bf16* __restrict__ ff1t) {
    int i = blockIdx.x * 256 + threadIdx.x;           // 4*2048*512
    int k = i & 511, n = (i >> 9) & 2047, l = i >> 20;
    ff1t[i] = f2bf(ff1_w[(size_t)l * Dsz * 4 * Dsz + (size_t)k * 4 * Dsz + n]);
}
__global__ void prep_ff2t(const float* __restrict__ ff2_w, __bf16* __restrict__ ff2t) {
    int i = blockIdx.x * 256 + threadIdx.x;           // 4*512*2048
    int k = i & 2047, n = (i >> 11) & 511, l = i >> 20;
    ff2t[i] = f2bf(ff2_w[(size_t)l * 4 * Dsz * Dsz + (size_t)k * Dsz + n]);
}

// ---------------- WMMA tile GEMM: [32 x 512(bf16,LDS)] x Wt[N][ldw](bf16) ----------------
// Each wave owns BOTH 16-row tiles and a 64-col slab (wid*64..+63, 4 tiles of 16),
// so every B fragment is loaded exactly once per WG and feeds two WMMAs.
template<bool RELU, bool ACC_TO_X, bool ADD_BIAS>
static __device__ __forceinline__ void gemm_tile(
    const __bf16* __restrict__ A, int lda,
    const __bf16* __restrict__ W, int ldw, int kOff,
    const float* __restrict__ bias,
    __bf16* __restrict__ OutB, float* __restrict__ OutX, int ldo,
    int tid)
{
    const int lane = tid & 31, wid = tid >> 5;
    const int cgB  = wid * 64;
    const int mrow = lane & 15;
    const int half = lane >> 4;
    const __bf16* arow0 = A + (size_t)mrow * lda + half * 8;          // rows 0..15
    const __bf16* arow1 = A + (size_t)(16 + mrow) * lda + half * 8;   // rows 16..31

    for (int ct = 0; ct < 4; ++ct) {
        const int ncol = cgB + ct * 16 + mrow;            // this lane's output column
        const __bf16* wrow = W + (size_t)ncol * ldw + kOff + half * 8;
        v8f acc0 = {}, acc1 = {};
        #pragma unroll
        for (int k0 = 0; k0 < Dsz; k0 += 32) {
            v8bf b0 = *(const v8bf*)(wrow + k0);          // K: k0+half*8 .. +7
            v8bf b1 = *(const v8bf*)(wrow + k0 + 16);     // K: k0+16+half*8 .. +7
            v8bf p0 = *(const v8bf*)(arow0 + k0);         // K: k0+half*8 .. +7
            v8bf p1 = *(const v8bf*)(arow0 + k0 + 16);    // K: k0+16+half*8 .. +7
            v8bf q0 = *(const v8bf*)(arow1 + k0);
            v8bf q1 = *(const v8bf*)(arow1 + k0 + 16);
            v16bf bv, a0v, a1v;
            #pragma unroll
            for (int i = 0; i < 8; ++i) {
                bv[i]  = b0[i]; bv[i + 8]  = b1[i];
                a0v[i] = p0[i]; a0v[i + 8] = p1[i];
                a1v[i] = q0[i]; a1v[i + 8] = q1[i];
            }
            acc0 = __builtin_amdgcn_wmma_f32_16x16x32_bf16(
                false, a0v, false, bv, (short)0, acc0, false, false);
            acc1 = __builtin_amdgcn_wmma_f32_16x16x32_bf16(
                false, a1v, false, bv, (short)0, acc1, false, false);
        }
        const float bn = ADD_BIAS ? bias[ncol] : 0.0f;
        #pragma unroll
        for (int r = 0; r < 8; ++r) {
            const int m = r + half * 8;                   // C layout: lanes16-31 -> M+8
            float v0 = acc0[r] + bn;
            float v1 = acc1[r] + bn;
            if (RELU) { v0 = fmaxf(v0, 0.0f); v1 = fmaxf(v1, 0.0f); }
            if (ACC_TO_X) {
                OutX[(size_t)m * LDX + ncol]        += v0;
                OutX[(size_t)(m + 16) * LDX + ncol] += v1;
            } else {
                OutB[(size_t)m * ldo + ncol]        = f2bf(v0);
                OutB[(size_t)(m + 16) * ldo + ncol] = f2bf(v1);
            }
        }
    }
}

// ---------------- layernorm: 8 threads per row, shuffle reduce ----------------
static __device__ __forceinline__ void layer_norm(
    const float* __restrict__ X, __bf16* __restrict__ Y,
    const float* __restrict__ g, const float* __restrict__ b, int tid)
{
    const int trow = tid >> 3, t8 = tid & 7, d0 = t8 * 64;
    const float* xr = X + (size_t)trow * LDX;
    float s = 0.f, ss = 0.f;
    #pragma unroll 8
    for (int j = 0; j < 64; ++j) { float v = xr[d0 + j]; s += v; ss += v * v; }
    s += __shfl_xor(s, 1, 32); ss += __shfl_xor(ss, 1, 32);
    s += __shfl_xor(s, 2, 32); ss += __shfl_xor(ss, 2, 32);
    s += __shfl_xor(s, 4, 32); ss += __shfl_xor(ss, 4, 32);
    const float mean = s * (1.0f / Dsz);
    const float var  = ss * (1.0f / Dsz) - mean * mean;
    const float rstd = rsqrtf(var + 1e-5f);
    __bf16* yr = Y + (size_t)trow * LDY;
    #pragma unroll 8
    for (int j = 0; j < 64; ++j) {
        const int d = d0 + j;
        yr[d] = f2bf((xr[d] - mean) * rstd * g[d] + b[d]);
    }
}

// ---------------- persistent scan kernel: 1 WG = 32 batch rows, all 64 steps ----------------
__global__ void __launch_bounds__(THREADS, 1)
scan_kernel(const int* __restrict__ a_seq, const int* __restrict__ b_seq,
            const float* __restrict__ bit_emb, const float* __restrict__ start_state,
            const float* __restrict__ ln1_g, const float* __restrict__ ln1_b,
            const float* __restrict__ qkv_b, const float* __restrict__ out_b,
            const float* __restrict__ ln2_g, const float* __restrict__ ln2_b,
            const float* __restrict__ ff1_b, const float* __restrict__ ff2_b,
            const float* __restrict__ head_w, const float* __restrict__ head_b,
            const __bf16* __restrict__ vwt, const __bf16* __restrict__ owt,
            const __bf16* __restrict__ ff1t, const __bf16* __restrict__ ff2t,
            float* __restrict__ out)
{
    __shared__ float  xs[ROWS][LDX];     // fp32 residual state
    __shared__ __bf16 yb[ROWS][LDY];     // ln output (GEMM A)
    __shared__ __bf16 vb[ROWS][LDY];     // v / ffn-hidden chunk (GEMM A)
    __shared__ float  embl[2][Dsz];      // VOCAB=2 embedding table
    __shared__ int    idxa[ROWS], idxb[ROWS];

    const int tid = threadIdx.x;
    const int r0  = blockIdx.x * ROWS;

    for (int i = tid; i < 2 * Dsz; i += THREADS) embl[i >> 9][i & 511] = bit_emb[i];
    for (int i = tid; i < ROWS * Dsz; i += THREADS)
        xs[i >> 9][i & 511] = start_state[i & 511];     // h0 broadcast
    __syncthreads();

    for (int s = 0; s < Ssz; ++s) {
        if (tid < ROWS) {
            idxa[tid] = a_seq[(size_t)(r0 + tid) * Ssz + s];
            idxb[tid] = b_seq[(size_t)(r0 + tid) * Ssz + s];
        }
        __syncthreads();
        // x = emb(a) + emb(b) + h
        for (int i = tid; i < ROWS * Dsz; i += THREADS) {
            const int r = i >> 9, d = i & 511;
            xs[r][d] += embl[idxa[r]][d] + embl[idxb[r]][d];
        }
        __syncthreads();

        for (int l = 0; l < DEPTH; ++l) {
            // --- attention branch (only V path is live) ---
            layer_norm(&xs[0][0], &yb[0][0], ln1_g + l * Dsz, ln1_b + l * Dsz, tid);
            __syncthreads();
            gemm_tile<false, false, true>(&yb[0][0], LDY,
                vwt + (size_t)l * Dsz * Dsz, Dsz, 0,
                qkv_b + (size_t)l * 3 * Dsz + 2 * Dsz, &vb[0][0], nullptr, LDY, tid);
            __syncthreads();
            gemm_tile<false, true, true>(&vb[0][0], LDY,
                owt + (size_t)l * Dsz * Dsz, Dsz, 0,
                out_b + (size_t)l * Dsz, nullptr, &xs[0][0], 0, tid);
            __syncthreads();

            // --- FFN, hidden dim tiled in 4 chunks of 512 ---
            layer_norm(&xs[0][0], &yb[0][0], ln2_g + l * Dsz, ln2_b + l * Dsz, tid);
            __syncthreads();
            for (int c = 0; c < 4; ++c) {
                gemm_tile<true, false, true>(&yb[0][0], LDY,
                    ff1t + ((size_t)l * 4 * Dsz + c * Dsz) * Dsz, Dsz, 0,
                    ff1_b + (size_t)l * 4 * Dsz + c * Dsz, &vb[0][0], nullptr, LDY, tid);
                __syncthreads();
                if (c == 0)
                    gemm_tile<false, true, true>(&vb[0][0], LDY,
                        ff2t + (size_t)l * Dsz * 4 * Dsz, 4 * Dsz, c * Dsz,
                        ff2_b + (size_t)l * Dsz, nullptr, &xs[0][0], 0, tid);
                else
                    gemm_tile<false, true, false>(&vb[0][0], LDY,
                        ff2t + (size_t)l * Dsz * 4 * Dsz, 4 * Dsz, c * Dsz,
                        nullptr, nullptr, &xs[0][0], 0, tid);
                __syncthreads();
            }
        }

        // --- head: logits[row, s, 0..1] = x . head_w + head_b ---
        {
            const int trow = tid >> 3, t8 = tid & 7, d0 = t8 * 64;
            float p0 = 0.f, p1 = 0.f;
            #pragma unroll 8
            for (int j = 0; j < 64; ++j) {
                const float xv = xs[trow][d0 + j];
                p0 += xv * head_w[(d0 + j) * 2 + 0];
                p1 += xv * head_w[(d0 + j) * 2 + 1];
            }
            p0 += __shfl_xor(p0, 1, 32); p1 += __shfl_xor(p1, 1, 32);
            p0 += __shfl_xor(p0, 2, 32); p1 += __shfl_xor(p1, 2, 32);
            p0 += __shfl_xor(p0, 4, 32); p1 += __shfl_xor(p1, 4, 32);
            if (t8 == 0) {
                const size_t o = ((size_t)(r0 + trow) * Ssz + s) * 2;
                out[o + 0] = p0 + head_b[0];
                out[o + 1] = p1 + head_b[1];
            }
        }
        __syncthreads();   // xs reused next step
    }
}

extern "C" void kernel_launch(void* const* d_in, const int* in_sizes, int n_in,
                              void* d_out, int out_size, void* d_ws, size_t ws_size,
                              hipStream_t stream) {
    const int*   a_seq  = (const int*)  d_in[0];
    const int*   b_seq  = (const int*)  d_in[1];
    const float* bitemb = (const float*)d_in[2];
    const float* sstate = (const float*)d_in[3];
    const float* ln1_g  = (const float*)d_in[4];
    const float* ln1_b  = (const float*)d_in[5];
    const float* qkv_w  = (const float*)d_in[6];
    const float* qkv_b  = (const float*)d_in[7];
    const float* out_w  = (const float*)d_in[8];
    const float* out_b  = (const float*)d_in[9];
    const float* ln2_g  = (const float*)d_in[10];
    const float* ln2_b  = (const float*)d_in[11];
    const float* ff1_w  = (const float*)d_in[12];
    const float* ff1_b  = (const float*)d_in[13];
    const float* ff2_w  = (const float*)d_in[14];
    const float* ff2_b  = (const float*)d_in[15];
    const float* head_w = (const float*)d_in[16];
    const float* head_b = (const float*)d_in[17];

    // workspace layout (bf16, transposed weights)
    char* ws = (char*)d_ws;
    __bf16* vwt  = (__bf16*)(ws);                         //  4*512*512   (2 MB)
    __bf16* owt  = (__bf16*)(ws + (size_t)2  * 1048576);  //  4*512*512   (2 MB)
    __bf16* ff1t = (__bf16*)(ws + (size_t)4  * 1048576);  //  4*2048*512  (8 MB)
    __bf16* ff2t = (__bf16*)(ws + (size_t)12 * 1048576);  //  4*512*2048  (8 MB)

    prep_vwt <<<(DEPTH * Dsz * Dsz)     / 256, 256, 0, stream>>>(qkv_w, vwt);
    prep_owt <<<(DEPTH * Dsz * Dsz)     / 256, 256, 0, stream>>>(out_w, owt);
    prep_ff1t<<<(DEPTH * Dsz * 4 * Dsz) / 256, 256, 0, stream>>>(ff1_w, ff1t);
    prep_ff2t<<<(DEPTH * 4 * Dsz * Dsz) / 256, 256, 0, stream>>>(ff2_w, ff2t);

    const int nrows = in_sizes[0] / Ssz;                  // B = 4096
    scan_kernel<<<nrows / ROWS, THREADS, 0, stream>>>(
        a_seq, b_seq, bitemb, sstate, ln1_g, ln1_b, qkv_b, out_b,
        ln2_g, ln2_b, ff1_b, ff2_b, head_w, head_b,
        vwt, owt, ff1t, ff2t, (float*)d_out);
}
